// Paired_CGCNN_weightshare_19009525252278
// MI455X (gfx1250) — compile-verified, matrix operand order checked by reading, not run on hardware
//
#include <hip/hip_runtime.h>
#include <hip/hip_bf16.h>
#include <cstdint>

#define NATOMS 51200
#define NM     12
#define FDIM   64
#define NBRF   41
#define NBRP   64           // nbr_fea row padded to 64 f16 (zero-filled)
#define KTOT   169          // 2F + NBR
#define KPAD   192          // 3 x 64 segments = 6 chunks of 32 for wmma f16
#define TWOF   128
#define NCRYS  128
#define APC    400
#define HDIM   128
#define EPS_BN 1e-5f

typedef _Float16 half8  __attribute__((ext_vector_type(8)));
typedef _Float16 half16 __attribute__((ext_vector_type(16)));
typedef float    floatx8 __attribute__((ext_vector_type(8)));
typedef int      int4v   __attribute__((ext_vector_type(4)));

union V16 { half16 v; half8 h[2]; };

#if defined(__has_builtin)
#if __has_builtin(__builtin_amdgcn_global_load_async_to_lds_b128)
#define ASYNC_LDS 1
#endif
#endif

// builtin expects v4i pointers in AS1 (global) / AS3 (LDS)
#define AS1V4(p) ((__attribute__((address_space(1))) int4v*)(p))
#define AS3V4(p) ((__attribute__((address_space(3))) int4v*)(p))

__device__ __forceinline__ float softplusf(float x){
    return (x > 20.f) ? x : log1pf(__expf(x));
}
__device__ __forceinline__ float sigmoidf_(float x){
    return 1.f / (1.f + __expf(-x));
}

// ---------------------------------------------------------------------------
// x = atom_fea @ embW + embb ; also write f16 copy for WMMA gathers
__global__ __launch_bounds__(256) void k_embed(
    const float* __restrict__ af, const float* __restrict__ eW,
    const float* __restrict__ eb, float* __restrict__ x, _Float16* __restrict__ xh)
{
    long i = (long)blockIdx.x * 256 + threadIdx.x;      // < N*64
    int n = (int)(i >> 6), f = (int)(i & 63);
    float acc = eb[f];
    const float* a = af + (long)n * 92;
    #pragma unroll 4
    for (int k = 0; k < 92; ++k) acc += a[k] * eW[k * FDIM + f];
    x[i]  = acc;
    xh[i] = (_Float16)acc;
}

// fp32 nbr_fea -> f16, rows padded 41 -> 64 with zeros (16B-aligned rows)
__global__ __launch_bounds__(256) void k_cvt_pad(
    const float* __restrict__ in, _Float16* __restrict__ out)
{
    long i = (long)blockIdx.x * 256 + threadIdx.x;      // < N*M*64
    long row = i >> 6; int c = (int)(i & 63);
    out[i] = (c < NBRF) ? (_Float16)in[row * NBRF + c] : (_Float16)0.f;
}

// Transpose+pad layer weights to [128][192] f16 ; zero the stats buffer (384 f32)
__global__ __launch_bounds__(256) void k_prep_w(
    const float* __restrict__ fcW, _Float16* __restrict__ wt, float* __restrict__ stats)
{
    int i = blockIdx.x * 256 + threadIdx.x;             // < 128*192
    int nn = i / KPAD, k = i % KPAD;
    wt[i] = (k < KTOT) ? (_Float16)fcW[k * TWOF + nn] : (_Float16)0.f;
    if (i < 384) stats[i] = 0.f;
}

// ---------------------------------------------------------------------------
// Main WMMA GEMM: gated[r][c] = concat(x[n], x[nbr], nbr_fea)[r] . W[:,c] + b[c]
// + fused BN1 statistics. Block = 256 thr (8 waves) -> 32 rows x 128 cols.
__global__ __launch_bounds__(256) void k_conv_mm(
    const _Float16* __restrict__ xh, const _Float16* __restrict__ nbrh,
    const int* __restrict__ nidx, const _Float16* __restrict__ wt,
    const float* __restrict__ bias, _Float16* __restrict__ gated,
    float* __restrict__ stats)
{
    __shared__ __align__(16) _Float16 sW[TWOF * KPAD];   // 48 KB
    __shared__ __align__(16) _Float16 sA[32 * KPAD];     // 12 KB (reused for D-tile)
    __shared__ float sSum[TWOF], sSq[TWOF];              // 1 KB

    const int tid  = threadIdx.x;
    const int lane = tid & 31;
    const int wave = tid >> 5;

    // ---- stage weights: 24576 halfs = 3072 x b128 (async if available) ----
#ifdef ASYNC_LDS
    {
        const char* g = (const char*)wt;
        char* l = (char*)sW;
        #pragma unroll
        for (int i = 0; i < 12; ++i) {
            __builtin_amdgcn_global_load_async_to_lds_b128(
                AS1V4(g + (size_t)(tid + i * 256) * 16),
                AS3V4(l + (size_t)(tid + i * 256) * 16), 0, 0);
        }
    }
#else
    {
        const half8* src = (const half8*)wt;
        half8* dst = (half8*)sW;
        #pragma unroll
        for (int i = 0; i < 12; ++i) dst[tid + i * 256] = src[tid + i * 256];
    }
#endif
    if (tid < TWOF) { sSum[tid] = 0.f; sSq[tid] = 0.f; }

    // ---- build A tile: 32 rows x 192, all-b128 gather/concat ----
    {
        const long rowBase = (long)blockIdx.x * 32;
        const int r = tid >> 3;                // 0..31
        const int q = tid & 7;                 // 0..7
        const long g = rowBase + r;
        const int n = (int)(g / NM);
        const int m = (int)(g - (long)n * NM);
        const int j = nidx[n * NM + m];
        const half8* xs = (const half8*)(xh + (long)n * FDIM);
        const half8* xn = (const half8*)(xh + (long)j * FDIM);
        const half8* nf = (const half8*)(nbrh + ((long)n * NM + m) * NBRP);
        half8* dst = (half8*)&sA[r * KPAD];
        dst[q]      = xs[q];
        dst[8 + q]  = xn[q];
        dst[16 + q] = nf[q];
    }
#ifdef ASYNC_LDS
#if defined(__has_builtin) && __has_builtin(__builtin_amdgcn_s_wait_asynccnt)
    __builtin_amdgcn_s_wait_asynccnt(0);
#else
    asm volatile("s_wait_asynccnt 0" ::: "memory");
#endif
#endif
    __syncthreads();

    const int rowTile  = wave & 1;     // which 16-row tile
    const int colGroup = wave >> 1;    // 0..3 -> cols [cg*32, cg*32+32)
    const int lr = lane & 15;
    const int lh = lane >> 4;

    floatx8 acc0, acc1;
    {
        float b0 = bias[colGroup * 32 + lr];
        float b1 = bias[colGroup * 32 + 16 + lr];
        #pragma unroll
        for (int i = 0; i < 8; ++i) { acc0[i] = b0; acc1[i] = b1; }
    }

    #pragma unroll
    for (int kc = 0; kc < 6; ++kc) {
        V16 a;
        {   // 16-bit A layout: lanes 0-15 -> K kc*32+{0..7,16..23}; lanes 16-31 -> +8
            const _Float16* p = &sA[(rowTile * 16 + lr) * KPAD + kc * 32 + lh * 8];
            a.h[0] = *(const half8*)p;
            a.h[1] = *(const half8*)(p + 16);
        }
        V16 b0, b1;
        {
            const _Float16* p0 = &sW[(colGroup * 32 + lr) * KPAD + kc * 32 + lh * 8];
            b0.h[0] = *(const half8*)p0;
            b0.h[1] = *(const half8*)(p0 + 16);
            const _Float16* p1 = p0 + 16 * KPAD;
            b1.h[0] = *(const half8*)p1;
            b1.h[1] = *(const half8*)(p1 + 16);
        }
        acc0 = __builtin_amdgcn_wmma_f32_16x16x32_f16(false, a.v, false, b0.v,
                                                      (short)0, acc0, false, false);
        acc1 = __builtin_amdgcn_wmma_f32_16x16x32_f16(false, a.v, false, b1.v,
                                                      (short)0, acc1, false, false);
    }

    // ---- BN1 stats: pre-reduce over the 8 rows held per lane (same feature) ----
    {
        const int c0 = colGroup * 32 + lr;
        const int c1 = c0 + 16;
        float s0 = 0.f, q0 = 0.f, s1 = 0.f, q1 = 0.f;
        #pragma unroll
        for (int i = 0; i < 8; ++i) {
            float v0 = acc0[i], v1 = acc1[i];
            s0 += v0; q0 += v0 * v0;
            s1 += v1; q1 += v1 * v1;
        }
        atomicAdd(&sSum[c0], s0); atomicAdd(&sSq[c0], q0);
        atomicAdd(&sSum[c1], s1); atomicAdd(&sSq[c1], q1);
    }

    // ---- stage D tile (32x128 f16) in LDS, then coalesced b128 stores ----
    __syncthreads();                       // all sA fragment reads finished
    {
        _Float16* sD = sA;                 // reuse 8 KB of sA
        const int c0 = colGroup * 32 + lr;
        const int c1 = c0 + 16;
        #pragma unroll
        for (int i = 0; i < 8; ++i) {
            int rrow = rowTile * 16 + (lh ? (8 + i) : i);
            sD[rrow * TWOF + c0] = (_Float16)acc0[i];
            sD[rrow * TWOF + c1] = (_Float16)acc1[i];
        }
    }
    __syncthreads();
    {
        const half8* s = (const half8*)sA;
        half8* gdst = (half8*)(gated + (long)blockIdx.x * 32 * TWOF);
        gdst[tid]       = s[tid];
        gdst[tid + 256] = s[tid + 256];
        if (tid < TWOF) {
            atomicAdd(&stats[tid],        sSum[tid]);
            atomicAdd(&stats[TWOF + tid], sSq[tid]);
        }
    }
}

// stats -> affine (scale, shift). 1 block, blockDim = feature count
__global__ void k_bn_final(const float* __restrict__ st, const float* __restrict__ g,
                           const float* __restrict__ b, float rows, float* __restrict__ out)
{
    int f = threadIdx.x, cnt = blockDim.x;
    float mean = st[f] / rows;
    float var  = st[cnt + f] / rows - mean * mean;
    float sc   = g[f] * rsqrtf(var + EPS_BN);
    out[f]       = sc;
    out[cnt + f] = b[f] - mean * sc;
}

// per-atom gating: sum_m sigmoid(BN(filter)) * softplus(BN(core))
// grid: N blocks x 64 threads; cooperative b128 load of the 12x128 tile
__global__ __launch_bounds__(64) void k_gate(
    const _Float16* __restrict__ gated, const float* __restrict__ bn1,
    float* __restrict__ summed)
{
    __shared__ __align__(16) _Float16 srow[NM * TWOF];   // 3 KB
    int n = blockIdx.x, f = threadIdx.x;
    {
        const half8* src = (const half8*)(gated + (long)n * NM * TWOF);
        half8* d = (half8*)srow;
        #pragma unroll
        for (int i = 0; i < 3; ++i) d[f + i * 64] = src[f + i * 64];
    }
    __syncthreads();
    float sF = bn1[f],        hF = bn1[TWOF + f];
    float sC = bn1[FDIM + f], hC = bn1[TWOF + FDIM + f];
    float acc = 0.f;
    #pragma unroll
    for (int m = 0; m < NM; ++m) {
        float fl = (float)srow[m * TWOF + f]        * sF + hF;
        float co = (float)srow[m * TWOF + FDIM + f] * sC + hC;
        acc += sigmoidf_(fl) * softplusf(co);
    }
    summed[(long)n * FDIM + f] = acc;
}

// BN2 statistics over summed (N x 64). grid-stride, LDS + global atomics.
__global__ __launch_bounds__(256) void k_bn2_stats(
    const float* __restrict__ summed, float* __restrict__ st)
{
    __shared__ float s[FDIM], q[FDIM];
    int tid = threadIdx.x;
    if (tid < FDIM) { s[tid] = 0.f; q[tid] = 0.f; }
    __syncthreads();
    const long total = (long)NATOMS * FDIM;
    for (long i = (long)blockIdx.x * blockDim.x + tid; i < total;
         i += (long)gridDim.x * blockDim.x) {
        float v = summed[i]; int f = (int)(i & 63);
        atomicAdd(&s[f], v); atomicAdd(&q[f], v * v);
    }
    __syncthreads();
    if (tid < FDIM) { atomicAdd(&st[tid], s[tid]); atomicAdd(&st[FDIM + tid], q[tid]); }
}

// x = softplus(x + BN2(summed)) ; refresh f16 copy
__global__ __launch_bounds__(256) void k_update(
    float* __restrict__ x, const float* __restrict__ summed,
    const float* __restrict__ bn2, _Float16* __restrict__ xh)
{
    long i = (long)blockIdx.x * 256 + threadIdx.x;
    int f = (int)(i & 63);
    float v = x[i] + summed[i] * bn2[f] + bn2[FDIM + f];
    float r = softplusf(v);
    x[i]  = r;
    xh[i] = (_Float16)r;
}

// pooled = mean over 400 atoms ; c = softplus(pooled @ denseW + denseb)
__global__ __launch_bounds__(128) void k_pool_dense(
    const float* __restrict__ x, const int* __restrict__ cidx,
    const float* __restrict__ dW, const float* __restrict__ db,
    float* __restrict__ cout)
{
    __shared__ float sp[FDIM];
    int c = blockIdx.x, tid = threadIdx.x;
    if (tid < FDIM) {
        float acc = 0.f;
        for (int a = 0; a < APC; ++a) {
            int at = cidx[c * APC + a];
            acc += x[(long)at * FDIM + tid];
        }
        sp[tid] = acc * (1.f / (float)APC);
    }
    __syncthreads();
    float acc = db[tid];
    #pragma unroll 8
    for (int k = 0; k < FDIM; ++k) acc += sp[k] * dW[k * HDIM + tid];
    cout[c * HDIM + tid] = softplusf(acc);
}

// head: out[c] = softplus(|cA-cB| @ ffW + ffb) @ outW + outb
__global__ __launch_bounds__(128) void k_head(
    const float* __restrict__ cA, const float* __restrict__ cB,
    const float* __restrict__ ffW, const float* __restrict__ ffb,
    const float* __restrict__ outW, const float* __restrict__ outb,
    float* __restrict__ out)
{
    __shared__ float sc[HDIM];
    __shared__ float sr[HDIM];
    int c = blockIdx.x, h = threadIdx.x;
    sc[h] = fabsf(cA[c * HDIM + h] - cB[c * HDIM + h]);
    __syncthreads();
    float acc = ffb[h];
    #pragma unroll 8
    for (int k = 0; k < HDIM; ++k) acc += sc[k] * ffW[k * HDIM + h];
    sr[h] = softplusf(acc) * outW[h];
    __syncthreads();
    for (int s = 64; s > 0; s >>= 1) {
        if (h < s) sr[h] += sr[h + s];
        __syncthreads();
    }
    if (h == 0) out[c] = sr[0] + outb[0];
}

// ---------------------------------------------------------------------------
extern "C" void kernel_launch(void* const* d_in, const int* in_sizes, int n_in,
                              void* d_out, int out_size, void* d_ws, size_t ws_size,
                              hipStream_t stream)
{
    (void)in_sizes; (void)n_in; (void)out_size; (void)ws_size;

    const float* embW   = (const float*)d_in[8];
    const float* embb   = (const float*)d_in[9];
    const float* fcW    = (const float*)d_in[10];
    const float* fcb    = (const float*)d_in[11];
    const float* bn1g   = (const float*)d_in[12];
    const float* bn1b   = (const float*)d_in[13];
    const float* bn2g   = (const float*)d_in[14];
    const float* bn2b   = (const float*)d_in[15];
    const float* denseW = (const float*)d_in[16];
    const float* denseb = (const float*)d_in[17];
    const float* ffW    = (const float*)d_in[18];
    const float* ffb    = (const float*)d_in[19];
    const float* outW   = (const float*)d_in[20];
    const float* outb   = (const float*)d_in[21];
    float* out = (float*)d_out;

    // workspace layout (256-byte aligned)
    char* ws = (char*)d_ws;
    size_t off = 0;
    auto take = [&](size_t bytes) -> char* {
        char* p = ws + off;
        off = (off + bytes + 255) & ~(size_t)255;
        return p;
    };
    float*    x      = (float*)   take((size_t)NATOMS * FDIM * 4);
    _Float16* xh     = (_Float16*)take((size_t)NATOMS * FDIM * 2);
    _Float16* nbrh   = (_Float16*)take((size_t)NATOMS * NM * NBRP * 2);
    _Float16* wt     = (_Float16*)take((size_t)TWOF * KPAD * 2);
    _Float16* gated  = (_Float16*)take((size_t)NATOMS * NM * TWOF * 2);
    float*    summed = (float*)   take((size_t)NATOMS * FDIM * 4);
    float*    stats  = (float*)   take(384 * 4);   // [sum1(128) sq1(128) sum2(64) sq2(64)]
    float*    bn1buf = (float*)   take(256 * 4);   // scale(128), shift(128)
    float*    bn2buf = (float*)   take(128 * 4);   // scale(64),  shift(64)
    float*    cA     = (float*)   take((size_t)NCRYS * HDIM * 4);
    float*    cB     = (float*)   take((size_t)NCRYS * HDIM * 4);

    const int gEmbed = (NATOMS * FDIM) / 256;          // 12800
    const int gCvt   = (NATOMS * NM * NBRP) / 256;     // 153600
    const int gConv  = (NATOMS * NM) / 32;             // 19200

    for (int br = 0; br < 2; ++br) {
        const float* af   = (const float*)d_in[br ? 4 : 0];
        const float* nbr  = (const float*)d_in[br ? 5 : 1];
        const int*   nidx = (const int*)  d_in[br ? 6 : 2];
        const int*   cidx = (const int*)  d_in[br ? 7 : 3];
        float*       cOut = br ? cB : cA;

        k_embed<<<gEmbed, 256, 0, stream>>>(af, embW, embb, x, xh);
        k_cvt_pad<<<gCvt, 256, 0, stream>>>(nbr, nbrh);

        for (int l = 0; l < 3; ++l) {
            k_prep_w<<<96, 256, 0, stream>>>(fcW + (size_t)l * KTOT * TWOF, wt, stats);
            k_conv_mm<<<gConv, 256, 0, stream>>>(xh, nbrh, nidx, wt,
                                                 fcb + l * TWOF, gated, stats);
            k_bn_final<<<1, TWOF, 0, stream>>>(stats, bn1g + l * TWOF, bn1b + l * TWOF,
                                               (float)(NATOMS * NM), bn1buf);
            k_gate<<<NATOMS, 64, 0, stream>>>(gated, bn1buf, summed);
            k_bn2_stats<<<512, 256, 0, stream>>>(summed, stats + 256);
            k_bn_final<<<1, FDIM, 0, stream>>>(stats + 256, bn2g + l * FDIM,
                                               bn2b + l * FDIM, (float)NATOMS, bn2buf);
            k_update<<<gEmbed, 256, 0, stream>>>(x, summed, bn2buf, xh);
        }
        k_pool_dense<<<NCRYS, 128, 0, stream>>>(x, cidx, denseW, denseb, cOut);
    }
    k_head<<<NCRYS, 128, 0, stream>>>(cA, cB, ffW, ffb, outW, outb, out);
}
